// EnhancedEmotionGNN_17386027615059
// MI455X (gfx1250) — matrix operand order repaired and checked by reading, not compiled, not run on hardware
//
#include <hip/hip_runtime.h>

// ---------------------------------------------------------------------------
// EnhancedEmotionGNN on MI455X (gfx1250): WMMA fp16 GEMMs + atomic GAT scatter
// GEMM v2: 128x64 block tile, 4 WMMA/wave/K-step, async global->LDS A staging
// ---------------------------------------------------------------------------

typedef __attribute__((ext_vector_type(16))) _Float16 v16h;
typedef __attribute__((ext_vector_type(8)))  float    v8f;
typedef long long ll;

#define HID 64

__device__ __forceinline__ float gelu_f(float x) {
    return 0.5f * x * (1.0f + erff(x * 0.70710678118654752f));
}
__device__ __forceinline__ unsigned short f2h_bits(float x) {
    union { _Float16 h; unsigned short u; } c; c.h = (_Float16)x; return c.u;
}
// order-preserving float<->uint map for atomicMax on floats
__device__ __forceinline__ unsigned fkey(float f) {
    unsigned u = __float_as_uint(f);
    return u ^ ((unsigned)((int)u >> 31) | 0x80000000u);
}
__device__ __forceinline__ float funkey(unsigned k) {
    unsigned u = (k & 0x80000000u) ? (k ^ 0x80000000u) : ~k;
    return __uint_as_float(u);
}

// ---------------------------------------------------------------------------
// WMMA GEMM: C[M,N] (f32) = act(A[M,K] (f16) @ B[K,N] (f16) + bias) [+ C]
// block = 256 threads (8 waves, 4x2), block tile 128(M) x 64(N), K step 32.
// Each wave owns a 2x2 grid of 16x16 WMMA tiles (4 wmma per K step).
// Requires: K%32==0, N%64==0, M%16==0 (true for all call sites).
// act: 0=none 1=gelu 2=tanh
// ---------------------------------------------------------------------------
__global__ __launch_bounds__(256)
void gemm_wmma_f16(const unsigned short* __restrict__ A,
                   const unsigned short* __restrict__ B,
                   const float* __restrict__ bias,
                   float* __restrict__ C,
                   int M, int N, int K, int act, int addC)
{
    __shared__ __align__(16) unsigned short As[128 * 32];  // [m][k]   8 KB
    __shared__ __align__(16) unsigned short Bt[64 * 32];   // [n][k]   4 KB

    const int t    = threadIdx.x;
    const int lane = t & 31;
    const int wave = t >> 5;
    const int wm   = wave >> 1;   // 0..3  -> rows wm*32 .. wm*32+31
    const int wn   = wave & 1;    // 0..1  -> cols wn*32 .. wn*32+31
    const int bm   = blockIdx.x * 128;
    const int bn   = blockIdx.y * 64;
    const int half = lane >> 4;   // 0/1
    const int l16  = lane & 15;

    // A loader: 16 halfs (32B) per thread
    const int la_row = t >> 1;          // 0..127
    const int la_col = (t & 1) * 16;    // 0 or 16
    // B loader: 8 halfs per thread, stored transposed
    const int lb_row = t >> 3;          // 0..31 (k)
    const int lb_col = (t & 7) * 8;     // 0..56 (n)

    const bool fullM = (bm + 128 <= M); // uniform per block

    v8f acc00 = {0.f,0.f,0.f,0.f,0.f,0.f,0.f,0.f};
    v8f acc01 = acc00, acc10 = acc00, acc11 = acc00;

    for (int kk = 0; kk < K; kk += 32) {
        // ---- stage A tile ----
        if (fullM) {
            // CDNA5 async global->LDS copy (ASYNCcnt), 2 x B128 per lane
            unsigned lds = (unsigned)(unsigned long long)
                           (const void*)(As + la_row * 32 + la_col);
            unsigned long long ga = (unsigned long long)
                           (A + (size_t)(bm + la_row) * K + kk + la_col);
            asm volatile("global_load_async_to_lds_b128 %0, %1, off"
                         :: "v"(lds), "v"(ga) : "memory");
            asm volatile("global_load_async_to_lds_b128 %0, %1, off"
                         :: "v"(lds + 16u), "v"(ga + 16ull) : "memory");
        } else {
            int gr = bm + la_row;
            uint4 v0, v1;
            v0.x = v0.y = v0.z = v0.w = 0u;
            v1 = v0;
            if (gr < M) {
                const uint4* gp = (const uint4*)(A + (size_t)gr * K + kk + la_col);
                v0 = gp[0];
                v1 = gp[1];
            }
            *(uint4*)(As + la_row * 32 + la_col)     = v0;
            *(uint4*)(As + la_row * 32 + la_col + 8) = v1;
        }
        // ---- stage B tile (transposed) ----
        {
            union { uint4 q; unsigned short u[8]; } bv;
            bv.q = *(const uint4*)(B + (size_t)(kk + lb_row) * N + bn + lb_col);
#pragma unroll
            for (int j = 0; j < 8; j++)
                Bt[(lb_col + j) * 32 + lb_row] = bv.u[j];
        }
        if (fullM) asm volatile("s_wait_asynccnt 0x0" ::: "memory");
        __syncthreads();

        // ---- fragments per documented gfx1250 VGPR layouts ----
        union { v16h v; uint4 q[2]; } fa0, fa1, fb0, fb1;
        // A 16x32: element e -> K = (e<8 ? 0 : 16) + half*8 + (e%8)
        const unsigned short* ap0 = As + (wm * 32 + l16) * 32;
        const unsigned short* ap1 = ap0 + 16 * 32;
        fa0.q[0] = *(const uint4*)(ap0 + half * 8);
        fa0.q[1] = *(const uint4*)(ap0 + 16 + half * 8);
        fa1.q[0] = *(const uint4*)(ap1 + half * 8);
        fa1.q[1] = *(const uint4*)(ap1 + 16 + half * 8);
        // B 32x16 (column n per lane): element e -> K = (e<8 ? 0 : 16) + half*8 + (e%8)
        const unsigned short* bq0 = Bt + (wn * 32 + l16) * 32 + half * 8;
        const unsigned short* bq1 = bq0 + 16 * 32;
        fb0.q[0] = *(const uint4*)(bq0);
        fb0.q[1] = *(const uint4*)(bq0 + 16);
        fb1.q[0] = *(const uint4*)(bq1);
        fb1.q[1] = *(const uint4*)(bq1 + 16);

        acc00 = __builtin_amdgcn_wmma_f32_16x16x32_f16(
                    false, fa0.v, false, fb0.v, (short)0, acc00, false, false);
        acc01 = __builtin_amdgcn_wmma_f32_16x16x32_f16(
                    false, fa0.v, false, fb1.v, (short)0, acc01, false, false);
        acc10 = __builtin_amdgcn_wmma_f32_16x16x32_f16(
                    false, fa1.v, false, fb0.v, (short)0, acc10, false, false);
        acc11 = __builtin_amdgcn_wmma_f32_16x16x32_f16(
                    false, fa1.v, false, fb1.v, (short)0, acc11, false, false);
        __syncthreads();
    }

    // ---- epilogue: D layout: VGPR i -> row = base + i + 8*half, col = l16 ----
#pragma unroll
    for (int mi = 0; mi < 2; mi++) {
#pragma unroll
        for (int ni = 0; ni < 2; ni++) {
            v8f acc = (mi == 0) ? ((ni == 0) ? acc00 : acc01)
                                : ((ni == 0) ? acc10 : acc11);
            const int col   = bn + wn * 32 + ni * 16 + l16;
            const int rbase = bm + wm * 32 + mi * 16 + half * 8;
            const float bv  = bias ? bias[col] : 0.f;
#pragma unroll
            for (int i = 0; i < 8; i++) {
                int r = rbase + i;
                if (r < M) {
                    float v = acc[i] + bv;
                    if (act == 1)      v = gelu_f(v);
                    else if (act == 2) v = tanhf(v);
                    float* cp = C + (size_t)r * N + col;
                    if (addC) v += *cp;
                    *cp = v;
                }
            }
        }
    }
}

// ---------------------------------------------------------------------------
// Utility / pointwise kernels
// ---------------------------------------------------------------------------
__global__ void zero_k(float* p, ll n) {
    for (ll i = (ll)blockIdx.x * blockDim.x + threadIdx.x; i < n;
         i += (ll)gridDim.x * blockDim.x) p[i] = 0.f;
}

__global__ void f2h_k(const float* __restrict__ s, unsigned short* __restrict__ d, ll n) {
    for (ll i = (ll)blockIdx.x * blockDim.x + threadIdx.x; i < n;
         i += (ll)gridDim.x * blockDim.x) d[i] = f2h_bits(s[i]);
}

// rowwise LayerNorm (in place), D <= 64
__global__ void ln_rows_k(float* __restrict__ X, const float* __restrict__ g,
                          const float* __restrict__ b, int M, int D) {
    int r = blockIdx.x * blockDim.x + threadIdx.x;
    if (r >= M) return;
    float* x = X + (size_t)r * D;
    float mu = 0.f;
    for (int i = 0; i < D; i++) mu += x[i];
    mu /= (float)D;
    float var = 0.f;
    for (int i = 0; i < D; i++) { float d = x[i] - mu; var += d * d; }
    var /= (float)D;
    float inv = rsqrtf(var + 1e-5f);
    for (int i = 0; i < D; i++) x[i] = (x[i] - mu) * inv * g[i] + b[i];
}

// xb (f16) = [ h[N,64] , emb[spk] [N,32] ]
__global__ void xb_build_k(const float* __restrict__ h, const float* __restrict__ emb,
                           const int* __restrict__ spk, unsigned short* __restrict__ xb,
                           int N) {
    ll i = (ll)blockIdx.x * blockDim.x + threadIdx.x;
    ll total = (ll)N * 96;
    if (i >= total) return;
    int n = (int)(i / 96), c = (int)(i % 96);
    float v = (c < 64) ? h[(size_t)n * 64 + c] : emb[(size_t)spk[n] * 32 + (c - 64)];
    xb[i] = f2h_bits(v);
}

// edge projection: ea_f[e,0:16] = LN(gelu(edge_attr[e,0:3] @ W + b))
__global__ void edge_proj_k(const float* __restrict__ ea3, const float* __restrict__ W,
                            const float* __restrict__ b, const float* __restrict__ g,
                            const float* __restrict__ be, float* __restrict__ out, int E) {
    int e = blockIdx.x * blockDim.x + threadIdx.x;
    if (e >= E) return;
    float a0 = ea3[(size_t)e * 3 + 0], a1 = ea3[(size_t)e * 3 + 1], a2 = ea3[(size_t)e * 3 + 2];
    float v[16];
    float mu = 0.f;
#pragma unroll
    for (int j = 0; j < 16; j++) {
        float s = a0 * W[0 * 16 + j] + a1 * W[1 * 16 + j] + a2 * W[2 * 16 + j] + b[j];
        s = gelu_f(s);
        v[j] = s; mu += s;
    }
    mu *= (1.f / 16.f);
    float var = 0.f;
#pragma unroll
    for (int j = 0; j < 16; j++) { float d = v[j] - mu; var += d * d; }
    var *= (1.f / 16.f);
    float inv = rsqrtf(var + 1e-5f);
#pragma unroll
    for (int j = 0; j < 16; j++)
        out[(size_t)e * 16 + j] = (v[j] - mu) * inv * g[j] + be[j];
}

__global__ void ea_mean_k(const float* __restrict__ ea, float* __restrict__ sum, int E) {
    __shared__ float ss[16];
    if (threadIdx.x < 16) ss[threadIdx.x] = 0.f;
    __syncthreads();
    int r = blockIdx.x * blockDim.x + threadIdx.x;
    if (r < E) {
#pragma unroll
        for (int j = 0; j < 16; j++) atomicAdd(&ss[j], ea[(size_t)r * 16 + j]);
    }
    __syncthreads();
    if (threadIdx.x < 16) atomicAdd(&sum[threadIdx.x], ss[threadIdx.x]);
}

__global__ void ea_loops_k(float* __restrict__ ea_f, const float* __restrict__ sum,
                           int E, int N) {
    ll i = (ll)blockIdx.x * blockDim.x + threadIdx.x;
    ll total = (ll)N * 16;
    if (i >= total) return;
    int j = (int)(i % 16);
    ea_f[(size_t)E * 16 + i] = sum[j] / (float)E;
}

__global__ void loops_idx_k(const int* __restrict__ src, const int* __restrict__ dst,
                            int* __restrict__ src2, int* __restrict__ dst2, int E, int N) {
    int i = blockIdx.x * blockDim.x + threadIdx.x;
    if (i >= E + N) return;
    if (i < E) { src2[i] = src[i]; dst2[i] = dst[i]; }
    else       { src2[i] = i - E;  dst2[i] = i - E; }
}

// al_s / al_d: per (n,h) reduction over C of xl * a
__global__ void al_reduce_k(const float* __restrict__ xl, const float* __restrict__ as,
                            const float* __restrict__ ad, float* __restrict__ al_s,
                            float* __restrict__ al_d, int N, int H, int C) {
    int i = blockIdx.x * blockDim.x + threadIdx.x;
    if (i >= N * H) return;
    int n = i / H, h = i % H;
    const float* x = xl + (size_t)n * H * C + (size_t)h * C;
    const float* pas = as + (size_t)h * C;
    const float* pad = ad + (size_t)h * C;
    float s = 0.f, d = 0.f;
    for (int c = 0; c < C; c++) { s += x[c] * pas[c]; d += x[c] * pad[c]; }
    al_s[i] = s; al_d[i] = d;
}

// fold a_edge into W_edge: Af[k,h] = sum_c We[k, h*C+c] * ae[h,c]
__global__ void fold_we_k(const float* __restrict__ We, const float* __restrict__ ae,
                          float* __restrict__ Af, int H, int C) {
    int i = blockIdx.x * blockDim.x + threadIdx.x;
    if (i >= 16 * H) return;
    int k = i / H, h = i % H;
    const float* w = We + (size_t)k * H * C + (size_t)h * C;
    const float* a = ae + (size_t)h * C;
    float s = 0.f;
    for (int c = 0; c < C; c++) s += w[c] * a[c];
    Af[(size_t)k * H + h] = s;
}

__global__ void al_e_k(const float* __restrict__ ea_f, const float* __restrict__ Af,
                       float* __restrict__ al_e, int E2, int H) {
    ll i = (ll)blockIdx.x * blockDim.x + threadIdx.x;
    if (i >= (ll)E2 * H) return;
    int e = (int)(i / H), h = (int)(i % H);
    const float* r = ea_f + (size_t)e * 16;
    float s = 0.f;
#pragma unroll
    for (int k = 0; k < 16; k++) s += r[k] * Af[k * H + h];
    al_e[i] = s;
}

// logits + leaky_relu(0.2) + fused segment-max (atomicMax on ordered keys)
__global__ void logits_k(const float* __restrict__ al_s, const float* __restrict__ al_d,
                         const float* __restrict__ al_e, const int* __restrict__ src2,
                         const int* __restrict__ dst2, float* __restrict__ lg,
                         unsigned* __restrict__ mkey, int E2, int H) {
    ll i = (ll)blockIdx.x * blockDim.x + threadIdx.x;
    if (i >= (ll)E2 * H) return;
    int e = (int)(i / H), h = (int)(i % H);
    float s = al_s[(size_t)src2[e] * H + h] + al_d[(size_t)dst2[e] * H + h] + al_e[i];
    s = (s > 0.f) ? s : 0.2f * s;
    lg[i] = s;
    atomicMax(&mkey[(size_t)dst2[e] * H + h], fkey(s));
}

// ex = exp(logit - segmax); accumulate denominator
__global__ void ex_k(float* __restrict__ lg, const unsigned* __restrict__ mkey,
                     float* __restrict__ den, const int* __restrict__ dst2, int E2, int H) {
    ll i = (ll)blockIdx.x * blockDim.x + threadIdx.x;
    if (i >= (ll)E2 * H) return;
    int e = (int)(i / H), h = (int)(i % H);
    float m = funkey(mkey[(size_t)dst2[e] * H + h]);
    float x = expf(lg[i] - m);
    lg[i] = x;
    atomicAdd(&den[(size_t)dst2[e] * H + h], x);
}

__global__ void rcp_k(float* __restrict__ den, ll n) {
    ll i = (ll)blockIdx.x * blockDim.x + threadIdx.x;
    if (i < n) den[i] = 1.f / fmaxf(den[i], 1e-16f);
}

// message scatter: agg[dst, h, c] += xl[src, h, c] * alpha
__global__ void scatter_k(const float* __restrict__ xl, const float* __restrict__ ex,
                          const float* __restrict__ rden, const int* __restrict__ src2,
                          const int* __restrict__ dst2, float* __restrict__ agg,
                          int E2, int H, int C) {
    ll total = (ll)E2 * H * C;
    int hc = H * C;
    for (ll i = (ll)blockIdx.x * blockDim.x + threadIdx.x; i < total;
         i += (ll)gridDim.x * blockDim.x) {
        int e = (int)(i / hc), rem = (int)(i % hc);
        int h = rem / C;
        int d = dst2[e];
        float v = xl[(size_t)src2[e] * hc + rem] * ex[(size_t)e * H + h]
                * rden[(size_t)d * H + h];
        atomicAdd(&agg[(size_t)d * hc + rem], v);
    }
}

// x = gelu(agg + bias) (concat heads), may be in place
__global__ void fin_concat_k(const float* __restrict__ agg, const float* __restrict__ bias,
                             float* __restrict__ x, int N, int HC) {
    ll i = (ll)blockIdx.x * blockDim.x + threadIdx.x;
    if (i >= (ll)N * HC) return;
    x[i] = gelu_f(agg[i] + bias[(int)(i % HC)]);
}

// x = gelu(mean_h(agg) + bias)
__global__ void fin_mean_k(const float* __restrict__ agg, const float* __restrict__ bias,
                           float* __restrict__ x, int N, int H, int C) {
    ll i = (ll)blockIdx.x * blockDim.x + threadIdx.x;
    if (i >= (ll)N * C) return;
    int n = (int)(i / C), c = (int)(i % C);
    float s = 0.f;
    for (int h = 0; h < H; h++) s += agg[(size_t)n * H * C + (size_t)h * C + c];
    x[i] = gelu_f(s / (float)H + bias[c]);
}

// small dense head: out[n,o] = X[n,:] @ W[:,o] + b[o]
__global__ void head_k(const float* __restrict__ X, const float* __restrict__ W,
                       const float* __restrict__ b, float* __restrict__ out,
                       int N, int Din, int Dout) {
    ll i = (ll)blockIdx.x * blockDim.x + threadIdx.x;
    if (i >= (ll)N * Dout) return;
    int n = (int)(i / Dout), o = (int)(i % Dout);
    const float* x = X + (size_t)n * Din;
    float s = 0.f;
    for (int k = 0; k < Din; k++) s += x[k] * W[(size_t)k * Dout + o];
    out[i] = s + b[o];
}

__global__ void maxkey_k(const float* __restrict__ s, unsigned* __restrict__ key, int N) {
    int i = blockIdx.x * blockDim.x + threadIdx.x;
    if (i < N) atomicMax(key, fkey(s[i]));
}

// softmax-weighted context: Z += e, ctx[c] += e * x3[n,c]
__global__ void pool_k(const float* __restrict__ s, const unsigned* __restrict__ smax,
                       const float* __restrict__ x3, float* __restrict__ Z,
                       float* __restrict__ ctx, int N) {
    int n = blockIdx.x * blockDim.x + threadIdx.x;
    if (n >= N) return;
    float e = expf(s[n] - funkey(*smax));
    atomicAdd(Z, e);
    const float* x = x3 + (size_t)n * HID;
    for (int c = 0; c < HID; c++) atomicAdd(&ctx[c], e * x[c]);
}

__global__ void ctxnorm_k(float* __restrict__ ctx, const float* __restrict__ Z) {
    int c = threadIdx.x;
    if (c < HID) ctx[c] = ctx[c] / (*Z);
}

// xf (f16) = [ x3[N,64] , broadcast ctx[64] ]
__global__ void xf_build_k(const float* __restrict__ x3, const float* __restrict__ ctx,
                           unsigned short* __restrict__ xf, int N) {
    ll i = (ll)blockIdx.x * blockDim.x + threadIdx.x;
    if (i >= (ll)N * 128) return;
    int n = (int)(i / 128), c = (int)(i % 128);
    float v = (c < 64) ? x3[(size_t)n * 64 + c] : ctx[c - 64];
    xf[i] = f2h_bits(v);
}

// ---------------------------------------------------------------------------
// Host orchestration
// ---------------------------------------------------------------------------
static inline ll cdiv_h(ll a, ll b) { return (a + b - 1) / b; }

extern "C" void kernel_launch(void* const* d_in, const int* in_sizes, int n_in,
                              void* d_out, int out_size, void* d_ws, size_t ws_size,
                              hipStream_t stream) {
    // input indices (setup_inputs dict order; params flattened in insertion order)
    enum {
        I_X = 0, I_EA, I_EMB, I_IPW, I_IPB, I_IPG, I_IPBE,
        I_EPW, I_EPB, I_EPG, I_EPBE,
        I_C1W, I_C1AS, I_C1AD, I_C1WE, I_C1AE, I_C1B, I_R1W, I_R1B,
        I_C2W, I_C2AS, I_C2AD, I_C2WE, I_C2AE, I_C2B, I_R2W, I_R2B,
        I_C3W, I_C3AS, I_C3AD, I_C3WE, I_C3AE, I_C3B,
        I_AXW1, I_AXB1, I_AXW2, I_AXB2,
        I_EAW1, I_EAB1, I_EAW2, I_EAB2,
        I_MW1, I_MB1, I_MG, I_MBE, I_MW2, I_MB2,
        I_SPK, I_EI
    };
    const float* x     = (const float*)d_in[I_X];
    const float* eattr = (const float*)d_in[I_EA];
    const int*   spk   = (const int*)d_in[I_SPK];
    const int*   ei    = (const int*)d_in[I_EI];

    const int N  = in_sizes[I_SPK];
    const int E  = in_sizes[I_EI] / 2;
    const int E2 = E + N;
    const int DIN = in_sizes[I_X] / N;  // 1024

    const int* src = ei;
    const int* dst = ei + E;

    // workspace carve-out
    char* ws = (char*)d_ws;
    size_t off = 0;
    auto carve = [&](size_t bytes) -> char* {
        char* p = ws + off; off = (off + bytes + 255) & ~(size_t)255; return p;
    };
    unsigned short* x_h    = (unsigned short*)carve((size_t)N * DIN * 2);
    unsigned short* xb_h   = (unsigned short*)carve((size_t)N * 96 * 2);
    unsigned short* x1_h   = (unsigned short*)carve((size_t)N * 512 * 2);
    unsigned short* x2_h   = (unsigned short*)carve((size_t)N * 384 * 2);
    unsigned short* x3_h   = (unsigned short*)carve((size_t)N * 64 * 2);
    unsigned short* xf_h   = (unsigned short*)carve((size_t)N * 128 * 2);
    unsigned short* ipW_h  = (unsigned short*)carve((size_t)DIN * 64 * 2);
    unsigned short* c1W_h  = (unsigned short*)carve((size_t)96 * 512 * 2);
    unsigned short* r1W_h  = (unsigned short*)carve((size_t)96 * 512 * 2);
    unsigned short* c2W_h  = (unsigned short*)carve((size_t)512 * 384 * 2);
    unsigned short* r2W_h  = (unsigned short*)carve((size_t)512 * 384 * 2);
    unsigned short* c3W_h  = (unsigned short*)carve((size_t)384 * 256 * 2);
    unsigned short* axW1_h = (unsigned short*)carve((size_t)384 * 128 * 2);
    unsigned short* eaW1_h = (unsigned short*)carve((size_t)64 * 64 * 2);
    unsigned short* mW1_h  = (unsigned short*)carve((size_t)128 * 64 * 2);
    float* bufA  = (float*)carve((size_t)N * 512 * 4);   // xl
    float* bufB  = (float*)carve((size_t)N * 512 * 4);   // agg / x_next
    float* h64   = (float*)carve((size_t)N * 64 * 4);    // ip output, reused for heads
    float* t128  = (float*)carve((size_t)N * 128 * 4);   // aux hidden
    float* x3f   = (float*)carve((size_t)N * 64 * 4);
    float* ea_f  = (float*)carve((size_t)E2 * 16 * 4);
    float* easum = (float*)carve(16 * 4);
    int*   src2  = (int*)carve((size_t)E2 * 4);
    int*   dst2  = (int*)carve((size_t)E2 * 4);
    float* al_s  = (float*)carve((size_t)N * 6 * 4);
    float* al_d  = (float*)carve((size_t)N * 6 * 4);
    float* al_e  = (float*)carve((size_t)E2 * 6 * 4);
    float* lg    = (float*)carve((size_t)E2 * 6 * 4);
    unsigned* mkey = (unsigned*)carve((size_t)N * 6 * 4);
    float* den   = (float*)carve((size_t)N * 6 * 4);
    float* Af    = (float*)carve(16 * 6 * 4);
    float* smaxZ = (float*)carve(2 * 4);    // [0]=smax key (as uint), [1]=Z
    float* ctx   = (float*)carve(64 * 4);
    float* sc    = (float*)carve((size_t)N * 4);
    (void)ws_size; (void)n_in; (void)out_size;

    float* out_main = (float*)d_out;          // [N,7]
    float* out_aux  = (float*)d_out + (size_t)N * 7;

    const dim3 B256(256);
    auto gs = [&](ll n) { return dim3((unsigned)cdiv_h(n, 256)); };

    auto gemm = [&](const unsigned short* A_, const unsigned short* B_,
                    const float* bias_, float* C_, int M_, int N_, int K_,
                    int act_, int addC_) {
        dim3 g((unsigned)cdiv_h(M_, 128), (unsigned)(N_ / 64));
        gemm_wmma_f16<<<g, B256, 0, stream>>>(A_, B_, bias_, C_, M_, N_, K_, act_, addC_);
    };
    auto cvt = [&](const float* s, unsigned short* d, ll n) {
        f2h_k<<<gs(n), B256, 0, stream>>>(s, d, n);
    };
    auto zero = [&](float* p, ll n) {
        zero_k<<<gs(n), B256, 0, stream>>>(p, n);
    };

    // ---- weight conversions (f32 -> f16) ----
    cvt(x, x_h, (ll)N * DIN);
    cvt((const float*)d_in[I_IPW],  ipW_h,  (ll)DIN * 64);
    cvt((const float*)d_in[I_C1W],  c1W_h,  96 * 512);
    cvt((const float*)d_in[I_R1W],  r1W_h,  96 * 512);
    cvt((const float*)d_in[I_C2W],  c2W_h,  512 * 384);
    cvt((const float*)d_in[I_R2W],  r2W_h,  512 * 384);
    cvt((const float*)d_in[I_C3W],  c3W_h,  384 * 256);
    cvt((const float*)d_in[I_AXW1], axW1_h, 384 * 128);
    cvt((const float*)d_in[I_EAW1], eaW1_h, 64 * 64);
    cvt((const float*)d_in[I_MW1],  mW1_h,  128 * 64);

    // ---- input projection: h = LN(gelu(x @ ip_W + b)); xb = [h, emb[spk]] ----
    gemm(x_h, ipW_h, (const float*)d_in[I_IPB], h64, N, 64, DIN, /*gelu*/1, 0);
    ln_rows_k<<<gs(N), B256, 0, stream>>>(h64, (const float*)d_in[I_IPG],
                                          (const float*)d_in[I_IPBE], N, 64);
    xb_build_k<<<gs((ll)N * 96), B256, 0, stream>>>(h64, (const float*)d_in[I_EMB],
                                                    spk, xb_h, N);

    // ---- edge features + self loops ----
    edge_proj_k<<<gs(E), B256, 0, stream>>>(eattr, (const float*)d_in[I_EPW],
        (const float*)d_in[I_EPB], (const float*)d_in[I_EPG],
        (const float*)d_in[I_EPBE], ea_f, E);
    zero(easum, 16);
    ea_mean_k<<<gs(E), B256, 0, stream>>>(ea_f, easum, E);
    ea_loops_k<<<gs((ll)N * 16), B256, 0, stream>>>(ea_f, easum, E, N);
    loops_idx_k<<<gs(E2), B256, 0, stream>>>(src, dst, src2, dst2, E, N);

    // ---- one GAT conv ----
    auto gat = [&](const unsigned short* xin_h, int Kin, const unsigned short* W_h,
                   int H, int C, int ias, int iad, int iwe, int iae, int ib,
                   bool concat, float* xout) {
        const int HC = H * C;
        gemm(xin_h, W_h, nullptr, bufA, N, HC, Kin, 0, 0);                       // xl
        al_reduce_k<<<gs((ll)N * H), B256, 0, stream>>>(bufA,
            (const float*)d_in[ias], (const float*)d_in[iad], al_s, al_d, N, H, C);
        fold_we_k<<<dim3(1), dim3(128), 0, stream>>>((const float*)d_in[iwe],
            (const float*)d_in[iae], Af, H, C);
        al_e_k<<<gs((ll)E2 * H), B256, 0, stream>>>(ea_f, Af, al_e, E2, H);
        zero((float*)mkey, (ll)N * H);
        zero(den, (ll)N * H);
        logits_k<<<gs((ll)E2 * H), B256, 0, stream>>>(al_s, al_d, al_e, src2, dst2,
                                                      lg, mkey, E2, H);
        ex_k<<<gs((ll)E2 * H), B256, 0, stream>>>(lg, mkey, den, dst2, E2, H);
        rcp_k<<<gs((ll)N * H), B256, 0, stream>>>(den, (ll)N * H);
        zero(bufB, (ll)N * HC);
        scatter_k<<<gs((ll)E2 * HC), B256, 0, stream>>>(bufA, lg, den, src2, dst2,
                                                        bufB, E2, H, C);
        if (concat)
            fin_concat_k<<<gs((ll)N * HC), B256, 0, stream>>>(bufB,
                (const float*)d_in[ib], xout, N, HC);
        else
            fin_mean_k<<<gs((ll)N * C), B256, 0, stream>>>(bufB,
                (const float*)d_in[ib], xout, N, H, C);
    };

    // conv1: in=96, H=4, C=128, concat -> 512 ; + residual xb @ r1_W
    gat(xb_h, 96, c1W_h, 4, 128, I_C1AS, I_C1AD, I_C1WE, I_C1AE, I_C1B, true, bufB);
    gemm(xb_h, r1W_h, (const float*)d_in[I_R1B], bufB, N, 512, 96, 0, /*addC*/1);
    cvt(bufB, x1_h, (ll)N * 512);

    // conv2: in=512, H=6, C=64, concat -> 384 ; + residual x1 @ r2_W
    gat(x1_h, 512, c2W_h, 6, 64, I_C2AS, I_C2AD, I_C2WE, I_C2AE, I_C2B, true, bufB);
    gemm(x1_h, r2W_h, (const float*)d_in[I_R2B], bufB, N, 384, 512, 0, /*addC*/1);
    cvt(bufB, x2_h, (ll)N * 384);

    // conv3: in=384, H=4, C=64, mean -> 64
    gat(x2_h, 384, c3W_h, 4, 64, I_C3AS, I_C3AD, I_C3WE, I_C3AE, I_C3B, false, x3f);
    cvt(x3f, x3_h, (ll)N * 64);

    // aux head: gelu(x2 @ aux_W1 + b1) @ aux_W2 + b2
    gemm(x2_h, axW1_h, (const float*)d_in[I_AXB1], t128, N, 128, 384, /*gelu*/1, 0);
    head_k<<<gs((ll)N * 7), B256, 0, stream>>>(t128, (const float*)d_in[I_AXW2],
        (const float*)d_in[I_AXB2], out_aux, N, 128, 7);

    // attention pooling: scores = tanh(x3 @ ea_W1 + b1) @ ea_W2 + b2
    gemm(x3_h, eaW1_h, (const float*)d_in[I_EAB1], h64, N, 64, 64, /*tanh*/2, 0);
    head_k<<<gs((ll)N), B256, 0, stream>>>(h64, (const float*)d_in[I_EAW2],
        (const float*)d_in[I_EAB2], sc, N, 64, 1);
    zero(smaxZ, 2);
    zero(ctx, 64);
    maxkey_k<<<gs(N), B256, 0, stream>>>(sc, (unsigned*)&smaxZ[0], N);
    pool_k<<<gs(N), B256, 0, stream>>>(sc, (const unsigned*)&smaxZ[0], x3f,
                                       &smaxZ[1], ctx, N);
    ctxnorm_k<<<dim3(1), dim3(64), 0, stream>>>(ctx, &smaxZ[1]);
    xf_build_k<<<gs((ll)N * 128), B256, 0, stream>>>(x3f, ctx, xf_h, N);

    // main head: LN(gelu(xf @ m_W1 + b1)) @ m_W2 + b2
    gemm(xf_h, mW1_h, (const float*)d_in[I_MB1], h64, N, 64, 128, /*gelu*/1, 0);
    ln_rows_k<<<gs(N), B256, 0, stream>>>(h64, (const float*)d_in[I_MG],
                                          (const float*)d_in[I_MBE], N, 64);
    head_k<<<gs((ll)N * 7), B256, 0, stream>>>(h64, (const float*)d_in[I_MW2],
        (const float*)d_in[I_MB2], out_main, N, 64, 7);
}
